// SensorAutoEncoder_72267119723145
// MI455X (gfx1250) — matrix operand order repaired and checked by reading, not compiled, not run on hardware
//
#include <hip/hip_runtime.h>
#include <hip/hip_bf16.h>

typedef __attribute__((ext_vector_type(16))) _Float16 v16h;
typedef __attribute__((ext_vector_type(8)))  _Float16 v8h;
typedef __attribute__((ext_vector_type(8)))  float    v8f;

#define NFEAT 43
#define TILEK 64          // padded K stride (f16 elements) of the per-wave LDS tile
#define WPB   8           // waves per block (wave32)

struct WPtrs { const float* W[10]; const float* B[10]; };

// ---------------------------------------------------------------------------
// Prep kernel: repack fp32 weights into f16 WMMA B-fragments + padded biases.
// Fragment layout (contiguous K order, dense f16 B 32x16): 32 lanes x 16 f16,
// lane-contiguous (lane*32 bytes). lane&15 = N column, lane>>4 selects
// K 0..15 vs 16..31 within the 32-wide K chunk, VGPR v holds K pair (2v,2v+1).
// ---------------------------------------------------------------------------
__global__ __launch_bounds__(256) void ae_prep(WPtrs p, _Float16* __restrict__ wfrag,
                                               float* __restrict__ biasbuf) {
  const int Kd[10]  = {43,43,43,43,43,21,43,43,43,43};
  const int Nd[10]  = {43,43,43,43,21,43,43,43,43,43};
  const int NKC[10] = {2,2,2,2,2,1,2,2,2,2};
  // cumulative 32-bit-word offsets, 256 words per fragment
  const int WOFF[11]= {0,1536,3072,4608,6144,7168,7936,9472,11008,12544,14080};

  int tid = blockIdx.x * blockDim.x + threadIdx.x;

  if (tid < 640) {                       // padded biases: 10 layers x 64 floats
    int li = tid >> 6, j = tid & 63;
    biasbuf[tid] = (j < Nd[li]) ? p.B[li][j] : 0.0f;
  }

  if (tid < 14080) {                     // one packed f16-pair per thread
    int li = 0;
    while (tid >= WOFF[li + 1]) ++li;
    int widx = tid - WOFF[li];
    int frag = widx >> 8;
    int rem  = widx & 255;
    int lane = rem >> 3;
    int v    = rem & 7;
    int nt   = frag / NKC[li];           // output N-tile
    int kc   = frag - nt * NKC[li];      // K chunk (32 wide)
    int k0 = kc * 32 + ((lane & 16) ? 16 : 0) + 2 * v;
    int n  = nt * 16 + (lane & 15);
    int K = Kd[li], N = Nd[li];
    const float* W = p.W[li];
    float f0 = (k0     < K && n < N) ? W[(size_t)k0 * N + n]       : 0.0f;
    float f1 = (k0 + 1 < K && n < N) ? W[(size_t)(k0 + 1) * N + n] : 0.0f;
    wfrag[2 * tid]     = (_Float16)f0;
    wfrag[2 * tid + 1] = (_Float16)f1;
  }
}

// ---------------------------------------------------------------------------
// Main kernel helpers
// ---------------------------------------------------------------------------
__device__ __forceinline__ float fast_tanh(float x) {
#if __has_builtin(__builtin_amdgcn_tanhf)
  return __builtin_amdgcn_tanhf(x);        // V_TANH_F32 (gfx1250 TRANS op)
#elif __has_builtin(__builtin_amdgcn_tanh_f32)
  return __builtin_amdgcn_tanh_f32(x);
#else
  return tanhf(x);
#endif
}

// A fragment (16x32 f16) from row-major LDS tile, contiguous K order (matches B):
// lane&15 = row M; lanes 0-15: K kb+0..15;
// lanes 16-31: K kb+16..31.
__device__ __forceinline__ v16h load_a(const _Float16* t, int lane, int kb) {
  const _Float16* base = t + (lane & 15) * TILEK + kb + ((lane & 16) ? 16 : 0);
  v8h lo = *(const v8h*)(base);            // ds_load_b128
  v8h hi = *(const v8h*)(base + 8);        // ds_load_b128
  return __builtin_shufflevector(lo, hi, 0,1,2,3,4,5,6,7,8,9,10,11,12,13,14,15);
}

// B fragment: lane-contiguous prepacked 32 bytes per lane (WGP$-resident).
__device__ __forceinline__ v16h load_b(const _Float16* fb, int lane) {
  const _Float16* base = fb + lane * 16;
  v8h lo = *(const v8h*)(base);            // global_load_b128
  v8h hi = *(const v8h*)(base + 8);        // global_load_b128
  return __builtin_shufflevector(lo, hi, 0,1,2,3,4,5,6,7,8,9,10,11,12,13,14,15);
}

// One dense layer: NT output 16-wide N tiles, NK 32-wide K chunks.
template<int NT, int NK, bool DO_TANH, bool LAST>
__device__ __forceinline__ void do_layer(_Float16* t, int lane,
    const _Float16* __restrict__ wf, const float* __restrict__ bias,
    float* __restrict__ out, long long rowbase, long long nrows) {
  // Previous layer's LDS stores must land before we read A fragments.
  // Per-wave DS is in-order; the wait + memory clobber also pins compiler order.
  asm volatile("s_wait_dscnt 0x0" ::: "memory");
  v16h a0 = load_a(t, lane, 0);
  v16h a1 = a0;
  if constexpr (NK == 2) a1 = load_a(t, lane, 32);

  const int n16  = lane & 15;
  const int hi16 = (lane >> 4) & 1;

#pragma unroll
  for (int nt = 0; nt < NT; ++nt) {
    const _Float16* fb = wf + nt * (NK * 512);     // 512 f16 per fragment
    v16h b0 = load_b(fb, lane);
    v8f c = {};
    c = __builtin_amdgcn_wmma_f32_16x16x32_f16(false, a0, false, b0,
                                               (short)0, c, false, false);
    if constexpr (NK == 2) {
      v16h b1 = load_b(fb + 512, lane);
      c = __builtin_amdgcn_wmma_f32_16x16x32_f16(false, a1, false, b1,
                                                 (short)0, c, false, false);
    }
    const int col = nt * 16 + n16;                 // output feature index
    const float bv = bias[col];
#pragma unroll
    for (int v = 0; v < 8; ++v) {
      float f = c[v] + bv;
      if constexpr (DO_TANH) f = fast_tanh(f);
      const int row = v + hi16 * 8;                // C/D layout: M = v (+8 hi half)
      if constexpr (!LAST) {
        t[row * TILEK + col] = (_Float16)f;        // stage next layer's A in f16
      } else {
        long long grow = rowbase + row;
        if (col < NFEAT && grow < nrows)
          __builtin_nontemporal_store(f, out + grow * NFEAT + col);
      }
    }
  }
}

// ---------------------------------------------------------------------------
// Main kernel: 8 waves/block, one 16-row tile per wave, all 10 layers fused.
// ---------------------------------------------------------------------------
__global__ __launch_bounds__(256) void ae_main(const float* __restrict__ x,
    const _Float16* __restrict__ wfrag, const float* __restrict__ biasbuf,
    float* __restrict__ out, long long nrows) {
  __shared__ __align__(16) _Float16 tile[WPB][16 * TILEK];   // 16 KB

  const int lane = threadIdx.x & 31;
  const int wave = threadIdx.x >> 5;
  const long long rowbase = ((long long)blockIdx.x * WPB + wave) * 16;
  if (rowbase >= nrows) return;
  _Float16* t = tile[wave];                // wave-private slice: no barriers needed

  // Stage x: 16 rows x 64 cols f16, cols >= 43 zeroed (K padding invariant).
#pragma unroll
  for (int r = 0; r < 16; ++r) {
    long long row = rowbase + r;
    if (row >= nrows) row = nrows - 1;     // safe over-read on ragged tail
    const float* xr = x + row * NFEAT;
    t[r * TILEK + lane] = (_Float16)__builtin_nontemporal_load(xr + lane);
    float v2 = (lane + 32 < NFEAT) ? __builtin_nontemporal_load(xr + lane + 32) : 0.0f;
    t[r * TILEK + 32 + lane] = (_Float16)v2;
  }

  // Fragment base offsets (f16 elements) and bias offsets (floats).
  do_layer<3,2,true ,false>(t, lane, wfrag + 0,     biasbuf + 0*64, out, rowbase, nrows); // L0 43->43 tanh
  do_layer<3,2,true ,false>(t, lane, wfrag + 3072,  biasbuf + 1*64, out, rowbase, nrows); // L1
  do_layer<3,2,true ,false>(t, lane, wfrag + 6144,  biasbuf + 2*64, out, rowbase, nrows); // L2
  do_layer<3,2,true ,false>(t, lane, wfrag + 9216,  biasbuf + 3*64, out, rowbase, nrows); // L3
  do_layer<2,2,false,false>(t, lane, wfrag + 12288, biasbuf + 4*64, out, rowbase, nrows); // L4 43->21 (no tanh)
  do_layer<3,1,true ,false>(t, lane, wfrag + 14336, biasbuf + 5*64, out, rowbase, nrows); // L5 21->43 tanh
  do_layer<3,2,true ,false>(t, lane, wfrag + 15872, biasbuf + 6*64, out, rowbase, nrows); // L6
  do_layer<3,2,true ,false>(t, lane, wfrag + 18944, biasbuf + 7*64, out, rowbase, nrows); // L7
  do_layer<3,2,true ,false>(t, lane, wfrag + 22016, biasbuf + 8*64, out, rowbase, nrows); // L8
  do_layer<3,2,false,true >(t, lane, wfrag + 25088, biasbuf + 9*64, out, rowbase, nrows); // L9 -> out
}

// ---------------------------------------------------------------------------
extern "C" void kernel_launch(void* const* d_in, const int* in_sizes, int n_in,
                              void* d_out, int out_size, void* d_ws, size_t ws_size,
                              hipStream_t stream) {
  const float* x = (const float*)d_in[0];
  WPtrs p;
  for (int i = 0; i < 10; ++i) {
    p.W[i] = (const float*)d_in[1 + 2 * i];
    p.B[i] = (const float*)d_in[2 + 2 * i];
  }
  _Float16* wfrag = (_Float16*)d_ws;                       // 56320 bytes
  float* biasbuf  = (float*)((char*)d_ws + 56320);         // 2560 bytes

  long long nrows = (long long)in_sizes[0] / NFEAT;        // 1048576

  ae_prep<<<55, 256, 0, stream>>>(p, wfrag, biasbuf);

  long long tiles = (nrows + (16LL * WPB) - 1) / (16LL * WPB);   // 8192
  ae_main<<<(unsigned)tiles, 256, 0, stream>>>(x, wfrag, biasbuf,
                                               (float*)d_out, nrows);
}
